// Block_38465727103079
// MI455X (gfx1250) — matrix operand order repaired and checked by reading, compiled, not run on hardware
//
#include <hip/hip_runtime.h>
#include <hip/hip_fp16.h>

typedef __attribute__((ext_vector_type(16))) _Float16 v16h;
typedef __attribute__((ext_vector_type(8)))  _Float16 v8h;
typedef __attribute__((ext_vector_type(8)))  float    v8f;

#define DIM    768
#define HEADS  12
#define WINSZ  14
#define NW     196      // tokens per window
#define NPAD   208      // 13*16
#define KPAD   224      // 7*32
#define DH     64
#define MLPH   3072
#define NWIN   25       // 5x5 windows per image
#define MROWS  39200    // 200 windows * 196 tokens
#define NTOK   32768    // 8*64*64
#define ATT_SCALE 0.125f
#define SMEM_BYTES (NPAD*NPAD*4 + NPAD*KPAD*2 + DH*KPAD*2)

// ---------- WMMA fragment helpers ----------
// A-matrix 16x32 f16 (ISA 7.12.2): lane m=l&15, half h=l>>4,
// halves 0..7 = A[m][k0+8h .. +7], halves 8..15 = A[m][k0+16+8h .. +7]
__device__ __forceinline__ v16h ldfragA(const _Float16* p) {
  v8h lo = *(const v8h*)(p);
  v8h hi = *(const v8h*)(p + 16);
  v16h r;
#pragma unroll
  for (int i = 0; i < 8; ++i) { r[i] = lo[i]; r[i + 8] = hi[i]; }
  return r;
}
// B-matrix 32x16 f16: lane n=l&15 holds column n; half h=l>>4,
// halves 0..7 = K k0+8h..+7, halves 8..15 = K k0+16+8h..+7 (same as A).
__device__ __forceinline__ v16h ldfragB(const _Float16* p) {
  v8h lo = *(const v8h*)(p);
  v8h hi = *(const v8h*)(p + 16);
  v16h r;
#pragma unroll
  for (int i = 0; i < 8; ++i) { r[i] = lo[i]; r[i + 8] = hi[i]; }
  return r;
}
__device__ __forceinline__ v8f wmma16(v16h a, v16h b, v8f c) {
  return __builtin_amdgcn_wmma_f32_16x16x32_f16(false, a, false, b, (short)0, c, false, false);
}
__device__ __forceinline__ float wred_sum(float v) {
#pragma unroll
  for (int o = 16; o > 0; o >>= 1) v += __shfl_xor(v, o, 32);
  return v;
}
__device__ __forceinline__ float wred_max(float v) {
#pragma unroll
  for (int o = 16; o > 0; o >>= 1) v = fmaxf(v, __shfl_xor(v, o, 32));
  return v;
}

// ---------- weight f32 -> f16 ----------
__global__ __launch_bounds__(256) void cvt_f16_kernel(const float* __restrict__ in,
                                                      _Float16* __restrict__ out, int n) {
  int i = blockIdx.x * 256 + threadIdx.x;
  if (i < n) out[i] = (_Float16)in[i];
}

// ---------- LN1 + window partition (pad 70x70 with zeros) ----------
__global__ __launch_bounds__(256) void ln_win_kernel(const float* __restrict__ x,
    const float* __restrict__ g, const float* __restrict__ bb,
    _Float16* __restrict__ out)
{
  int wave = threadIdx.x >> 5, lane = threadIdx.x & 31;
  int row  = blockIdx.x * 8 + wave;           // [0, MROWS)
  int bw = row / NW, t = row - bw * NW;
  int b  = bw / NWIN, wi = bw - b * NWIN;
  int wy = wi / 5, wx = wi - wy * 5;
  int y  = wy * WINSZ + t / WINSZ;
  int xx = wx * WINSZ + (t - (t / WINSZ) * WINSZ);
  _Float16* orow = out + (size_t)row * DIM;
  if (y >= 64 || xx >= 64) {
    for (int c = lane; c < DIM; c += 32) orow[c] = (_Float16)0.0f;
    return;
  }
  const float* xr = x + ((size_t)b * 4096 + y * 64 + xx) * DIM;
  float v[24]; float s = 0.f, s2 = 0.f;
#pragma unroll
  for (int i = 0; i < 24; ++i) {
    float u = xr[lane + i * 32];
    v[i] = u; s += u; s2 += u * u;
  }
  s = wred_sum(s); s2 = wred_sum(s2);
  float mean = s * (1.0f / DIM);
  float rstd = rsqrtf(s2 * (1.0f / DIM) - mean * mean + 1e-5f);
#pragma unroll
  for (int i = 0; i < 24; ++i) {
    int c = lane + i * 32;
    orow[c] = (_Float16)((v[i] - mean) * rstd * g[c] + bb[c]);
  }
}

// ---------- LN2 ----------
__global__ __launch_bounds__(256) void ln2_kernel(const float* __restrict__ x1,
    const float* __restrict__ g, const float* __restrict__ bb,
    _Float16* __restrict__ out)
{
  int wave = threadIdx.x >> 5, lane = threadIdx.x & 31;
  int row  = blockIdx.x * 8 + wave;           // [0, NTOK)
  const float* xr = x1 + (size_t)row * DIM;
  _Float16* orow  = out + (size_t)row * DIM;
  float v[24]; float s = 0.f, s2 = 0.f;
#pragma unroll
  for (int i = 0; i < 24; ++i) {
    float u = xr[lane + i * 32];
    v[i] = u; s += u; s2 += u * u;
  }
  s = wred_sum(s); s2 = wred_sum(s2);
  float mean = s * (1.0f / DIM);
  float rstd = rsqrtf(s2 * (1.0f / DIM) - mean * mean + 1e-5f);
#pragma unroll
  for (int i = 0; i < 24; ++i) {
    int c = lane + i * 32;
    orow[c] = (_Float16)((v[i] - mean) * rstd * g[c] + bb[c]);
  }
}

// ---------- blocked WMMA GEMM: C[M,N] = A[M,K] @ W[N,K]^T + bias ----------
// Each wave computes a 32x64 macro-tile (2x4 grid of 16x16 WMMA tiles):
// per K-step: 2 A-frags + 4 B-frags feed 8 v_wmma -> 21 flops/byte from cache.
// EPI 0: f16 out (+bias)                    -> outh   (QKV)
// EPI 1: f16 out (+bias, exact GELU)        -> outh   (FC1)
// EPI 2: revert-window scatter + residual   -> outf   (PROJ)
// EPI 3: f32 out + residual                 -> outf   (FC2)
template <int EPI>
__global__ __launch_bounds__(256) void gemm_kernel(
    const _Float16* __restrict__ A, const _Float16* __restrict__ W,
    const float* __restrict__ bias, int K, int tilesN, int totalTiles, int N,
    _Float16* __restrict__ outh, float* __restrict__ outf,
    const float* __restrict__ res)
{
  int wave = threadIdx.x >> 5;
  int lane = threadIdx.x & 31;
  int flat = blockIdx.x * 8 + wave;
  if (flat >= totalTiles) return;
  int mt = flat / tilesN, nt = flat - mt * tilesN;
  int row0 = mt << 5, col0 = nt << 6;        // 32-row x 64-col macro-tile
  int m = lane & 15, hh = lane >> 4;

  const _Float16* a0 = A + (size_t)(row0 + m) * K + 8 * hh;
  const _Float16* b0 = W + (size_t)(col0 + m) * K + 8 * hh;
  size_t aStep = (size_t)16 * K;             // next 16-row tile
  v8f acc[2][4] = {};
#pragma unroll 2
  for (int k = 0; k < K; k += 32) {
    v16h af0 = ldfragA(a0 + k);
    v16h af1 = ldfragA(a0 + aStep + k);
    v16h bf0 = ldfragB(b0 + k);
    v16h bf1 = ldfragB(b0 + aStep + k);
    v16h bf2 = ldfragB(b0 + 2 * aStep + k);
    v16h bf3 = ldfragB(b0 + 3 * aStep + k);
    acc[0][0] = wmma16(af0, bf0, acc[0][0]);
    acc[0][1] = wmma16(af0, bf1, acc[0][1]);
    acc[0][2] = wmma16(af0, bf2, acc[0][2]);
    acc[0][3] = wmma16(af0, bf3, acc[0][3]);
    acc[1][0] = wmma16(af1, bf0, acc[1][0]);
    acc[1][1] = wmma16(af1, bf1, acc[1][1]);
    acc[1][2] = wmma16(af1, bf2, acc[1][2]);
    acc[1][3] = wmma16(af1, bf3, acc[1][3]);
  }

#pragma unroll
  for (int j = 0; j < 4; ++j) {
    int col = col0 + j * 16 + m;             // C/D layout: lane -> column n = lane&15
    float bn = bias[col];
#pragma unroll
    for (int i = 0; i < 2; ++i) {
#pragma unroll
      for (int p = 0; p < 8; ++p) {
        int r = row0 + i * 16 + p + 8 * hh;  // C/D layout: vgpr p -> row p+8h
        float v = acc[i][j][p] + bn;
        if (EPI == 0) {
          outh[(size_t)r * N + col] = (_Float16)v;
        } else if (EPI == 1) {
          v = 0.5f * v * (1.0f + erff(v * 0.70710678f));
          outh[(size_t)r * N + col] = (_Float16)v;
        } else if (EPI == 2) {
          int bw = r / NW,    t  = r - bw * NW;
          int b  = bw / NWIN, wi = bw - b * NWIN;
          int wy = wi / 5,    wx = wi - wy * 5;
          int iy = t / WINSZ, ix = t - iy * WINSZ;
          int y  = wy * WINSZ + iy, xx = wx * WINSZ + ix;
          if (y < 64 && xx < 64) {
            size_t o = ((size_t)b * 4096 + y * 64 + xx) * DIM + col;
            outf[o] = res[o] + v;
          }
        } else {
          size_t o = (size_t)r * DIM + col;
          outf[o] = res[o] + v;
        }
      }
    }
  }
}

// ---------- windowed attention: one block per (window, head) ----------
// LDS: S f32 [208][208] | P f16 [208][224] | Vt f16 [64][224]  = 294912 B < 320KB WGP LDS
__global__ __launch_bounds__(256) void attn_kernel(
    const _Float16* __restrict__ qkv,      // [MROWS][2304] = [.., (q|k|v), head, dh]
    const float* __restrict__ relb,        // [HEADS][196][196]
    _Float16* __restrict__ out)            // [MROWS][768]
{
  extern __shared__ __align__(16) char smem[];
  float*    S  = (float*)smem;                               // [NPAD][NPAD]
  _Float16* P  = (_Float16*)(smem + NPAD * NPAD * 4);        // [NPAD][KPAD]
  _Float16* Vt = P + NPAD * KPAD;                            // [DH][KPAD]

  int bw = blockIdx.x / HEADS;
  int h  = blockIdx.x - bw * HEADS;
  const _Float16* qb = qkv + (size_t)bw * NW * 2304 + h * DH;
  const _Float16* kb = qb + DIM;
  const _Float16* vb = qb + 2 * DIM;

  int wave = threadIdx.x >> 5, lane = threadIdx.x & 31;
  int m = lane & 15, hh = lane >> 4;

  // Phase 1: S = (q @ k^T) * scale. Wave owns row-strip ti; q-frags reused over 13 tj.
  for (int ti = wave; ti < 13; ti += 8) {
    int qr = ti * 16 + m; if (qr > 195) qr = 195;
    const _Float16* qp = qb + (size_t)qr * 2304 + 8 * hh;
    v16h aq0 = ldfragA(qp);
    v16h aq1 = ldfragA(qp + 32);
    for (int tj = 0; tj < 13; ++tj) {
      int kr = tj * 16 + m; if (kr > 195) kr = 195;
      const _Float16* kp = kb + (size_t)kr * 2304 + 8 * hh;
      v8f acc = {};
      acc = wmma16(aq0, ldfragB(kp),      acc);
      acc = wmma16(aq1, ldfragB(kp + 32), acc);
#pragma unroll
      for (int p = 0; p < 8; ++p)
        S[(ti * 16 + p + 8 * hh) * NPAD + tj * 16 + m] = acc[p] * ATT_SCALE;
    }
  }
  // Phase 1b: stage V^T into LDS (zero-padded K)
  for (int idx = threadIdx.x; idx < DH * KPAD; idx += 256) {
    int n = idx / KPAD, k = idx - n * KPAD;
    Vt[idx] = (k < NW) ? vb[(size_t)k * 2304 + n] : (_Float16)0.0f;
  }
  __syncthreads();

  // Phase 2: softmax rows (add rel-pos bias), write P (f16, zero-padded to KPAD)
  for (int r = wave; r < NW; r += 8) {
    const float* brow = relb + ((size_t)h * NW + r) * NW;
    const float* srow = S + (size_t)r * NPAD;
    float vals[7]; float mx = -1e30f;
#pragma unroll
    for (int i = 0; i < 7; ++i) {
      int c = lane + i * 32;
      float v = (c < NW) ? (srow[c] + brow[c]) : -1e30f;
      vals[i] = v; mx = fmaxf(mx, v);
    }
    mx = wred_max(mx);
    float sum = 0.f;
#pragma unroll
    for (int i = 0; i < 7; ++i) {
      int c = lane + i * 32;
      float e = (c < NW) ? __expf(vals[i] - mx) : 0.f;
      vals[i] = e; sum += e;
    }
    sum = wred_sum(sum);
    float inv = 1.0f / sum;
    _Float16* prow = P + (size_t)r * KPAD;
#pragma unroll
    for (int i = 0; i < 7; ++i) {
      int c = lane + i * 32;                 // covers 0..223 exactly
      prow[c] = (c < NW) ? (_Float16)(vals[i] * inv) : (_Float16)0.0f;
    }
  }
  __syncthreads();

  // Phase 3: out = P @ V. Wave owns row-strip ti; P-frag shared across 4 N-tiles.
  for (int ti = wave; ti < 13; ti += 8) {
    int pr = ti * 16 + m; if (pr > 195) pr = 195;  // pad rows: harmless, not stored
    const _Float16* ap = P + (size_t)pr * KPAD + 8 * hh;
    const _Float16* bp = Vt + (size_t)m * KPAD + 8 * hh;
    v8f acc[4] = {};
#pragma unroll
    for (int k = 0; k < KPAD; k += 32) {
      v16h af = ldfragA(ap + k);
      acc[0] = wmma16(af, ldfragB(bp + k),                 acc[0]);
      acc[1] = wmma16(af, ldfragB(bp + 16 * KPAD + k),     acc[1]);
      acc[2] = wmma16(af, ldfragB(bp + 32 * KPAD + k),     acc[2]);
      acc[3] = wmma16(af, ldfragB(bp + 48 * KPAD + k),     acc[3]);
    }
#pragma unroll
    for (int j = 0; j < 4; ++j) {
#pragma unroll
      for (int p = 0; p < 8; ++p) {
        int row = ti * 16 + p + 8 * hh;
        if (row < NW)
          out[((size_t)bw * NW + row) * DIM + h * DH + j * 16 + m] = (_Float16)acc[j][p];
      }
    }
  }
}

// ---------- host ----------
extern "C" void kernel_launch(void* const* d_in, const int* in_sizes, int n_in,
                              void* d_out, int out_size, void* d_ws, size_t ws_size,
                              hipStream_t stream)
{
  (void)in_sizes; (void)n_in; (void)out_size; (void)ws_size;
  const float* x      = (const float*)d_in[0];
  const float* relb   = (const float*)d_in[1];
  const float* ln1_g  = (const float*)d_in[2];
  const float* ln1_b  = (const float*)d_in[3];
  const float* qkv_w  = (const float*)d_in[4];
  const float* qkv_b  = (const float*)d_in[5];
  const float* proj_w = (const float*)d_in[6];
  const float* proj_b = (const float*)d_in[7];
  const float* ln2_g  = (const float*)d_in[8];
  const float* ln2_b  = (const float*)d_in[9];
  const float* fc1_w  = (const float*)d_in[10];
  const float* fc1_b  = (const float*)d_in[11];
  const float* fc2_w  = (const float*)d_in[12];
  const float* fc2_b  = (const float*)d_in[13];
  float* out = (float*)d_out;

  char* ws = (char*)d_ws;
  size_t off = 0;
  auto alloc = [&](size_t bytes) -> char* {
    char* p = ws + off;
    off += (bytes + 255) & ~(size_t)255;
    return p;
  };
  _Float16* wqkv  = (_Float16*)alloc((size_t)3 * DIM * DIM * 2);
  _Float16* wproj = (_Float16*)alloc((size_t)DIM * DIM * 2);
  _Float16* wfc1  = (_Float16*)alloc((size_t)MLPH * DIM * 2);
  _Float16* wfc2  = (_Float16*)alloc((size_t)DIM * MLPH * 2);
  _Float16* hwb   = (_Float16*)alloc((size_t)MROWS * DIM * 2);
  _Float16* qkvb  = (_Float16*)alloc((size_t)MROWS * 3 * DIM * 2);
  _Float16* attnb = (_Float16*)alloc((size_t)MROWS * DIM * 2);
  float*    x1    = (float*)   alloc((size_t)NTOK * DIM * 4);
  _Float16* ln2h  = (_Float16*)alloc((size_t)NTOK * DIM * 2);
  _Float16* h1    = (_Float16*)alloc((size_t)NTOK * MLPH * 2);

  // weights f32 -> f16 (layout already [N][K])
  cvt_f16_kernel<<<(3 * DIM * DIM + 255) / 256, 256, 0, stream>>>(qkv_w, wqkv, 3 * DIM * DIM);
  cvt_f16_kernel<<<(DIM * DIM + 255) / 256, 256, 0, stream>>>(proj_w, wproj, DIM * DIM);
  cvt_f16_kernel<<<(MLPH * DIM + 255) / 256, 256, 0, stream>>>(fc1_w, wfc1, MLPH * DIM);
  cvt_f16_kernel<<<(DIM * MLPH + 255) / 256, 256, 0, stream>>>(fc2_w, wfc2, DIM * MLPH);

  // LN1 + windowing
  ln_win_kernel<<<MROWS / 8, 256, 0, stream>>>(x, ln1_g, ln1_b, hwb);

  // QKV GEMM: [39200,768] x [2304,768]^T -> f16 [39200,2304]
  {
    int tilesM = MROWS / 32, tilesN = (3 * DIM) / 64;        // 1225 x 36
    int total = tilesM * tilesN;
    gemm_kernel<0><<<(total + 7) / 8, 256, 0, stream>>>(hwb, wqkv, qkv_b, DIM,
                                                        tilesN, total, 3 * DIM,
                                                        qkvb, nullptr, nullptr);
  }
  // attention per (window, head)
  attn_kernel<<<200 * HEADS, 256, SMEM_BYTES, stream>>>(qkvb, relb, attnb);

  // PROJ GEMM + revert window + residual -> x1 f32
  {
    int tilesM = MROWS / 32, tilesN = DIM / 64;              // 1225 x 12
    int total = tilesM * tilesN;
    gemm_kernel<2><<<(total + 7) / 8, 256, 0, stream>>>(attnb, wproj, proj_b, DIM,
                                                        tilesN, total, DIM,
                                                        nullptr, x1, x);
  }
  // LN2
  ln2_kernel<<<NTOK / 8, 256, 0, stream>>>(x1, ln2_g, ln2_b, ln2h);

  // FC1 + GELU -> f16 [32768,3072]
  {
    int tilesM = NTOK / 32, tilesN = MLPH / 64;              // 1024 x 48
    int total = tilesM * tilesN;
    gemm_kernel<1><<<(total + 7) / 8, 256, 0, stream>>>(ln2h, wfc1, fc1_b, DIM,
                                                        tilesN, total, MLPH,
                                                        h1, nullptr, nullptr);
  }
  // FC2 + residual -> d_out f32
  {
    int tilesM = NTOK / 32, tilesN = DIM / 64;               // 1024 x 12
    int total = tilesM * tilesN;
    gemm_kernel<3><<<(total + 7) / 8, 256, 0, stream>>>(h1, wfc2, fc2_b, MLPH,
                                                        tilesN, total, DIM,
                                                        nullptr, out, x1);
  }
}
